// MakeGeneratorLstm_33165737460018
// MI455X (gfx1250) — compile-verified, matrix operand order checked by reading, not run on hardware
//
#include <hip/hip_runtime.h>
#include <hip/hip_bf16.h>
#include <cstdint>
#include <cstddef>

// Problem dims (compile-time constants from the reference)
static constexpr int Nn  = 8;     // ensemble
static constexpr int Bb  = 64;    // batch
static constexpr int Tt  = 256;   // time
static constexpr int INd = 64;    // input dim
static constexpr int Hh  = 512;   // hidden
static constexpr int Mm  = 512;   // head mid
static constexpr int Oo  = 7;     // head out
static constexpr int G4H = 4 * Hh; // 2048 gates

typedef __hip_bfloat16 bh;
typedef __bf16        bf16x16 __attribute__((ext_vector_type(16)));
typedef float         f32x8   __attribute__((ext_vector_type(8)));
typedef unsigned int  u32x4   __attribute__((ext_vector_type(4)));

#define DEVFN __device__ __forceinline__

// ---------------------------------------------------------------------------
// WMMA helpers: CDNA5 v_wmma_f32_16x16x32_bf16, layouts per ISA 7.12.2.
// 16-bit A fragment for lane L (row = tile_row + (L&15), half = L>>4):
//   element e in [0,8)  -> k = k0 + half*8 + e
//   element e in [8,16) -> k = k0 + 16 + half*8 + (e-8)
// i.e. two contiguous 16-byte (8 x bf16) loads. B uses the symmetric layout
// with column-per-lane, reading rows of the row-major weight matrix (B = W^T).
// ---------------------------------------------------------------------------
DEVFN bf16x16 load_frag(const bh* rowptr, int k0, int half) {
  union { bf16x16 v; u32x4 q[2]; } f;
  f.q[0] = *reinterpret_cast<const u32x4*>(rowptr + k0 + half * 8);
  f.q[1] = *reinterpret_cast<const u32x4*>(rowptr + k0 + 16 + half * 8);
  return f.v;
}

DEVFN f32x8 zero8() {
  f32x8 v;
#pragma unroll
  for (int i = 0; i < 8; ++i) v[i] = 0.0f;
  return v;
}

DEVFN f32x8 wmma_bf16(bf16x16 a, bf16x16 b, f32x8 c) {
  return __builtin_amdgcn_wmma_f32_16x16x32_bf16(
      /*neg_a=*/false, a, /*neg_b=*/false, b,
      /*c_mod=*/(short)0, c, /*reuse_a=*/false, /*reuse_b=*/false);
}

DEVFN float sigm(float x) { return 1.0f / (1.0f + __expf(-x)); }

// ---------------------------------------------------------------------------
// Utility kernels
// ---------------------------------------------------------------------------
__global__ void k_f32_to_bf16(const float* __restrict__ s, bh* __restrict__ d, int n) {
  int i = blockIdx.x * blockDim.x + threadIdx.x;
  int stride = gridDim.x * blockDim.x;
  for (; i < n; i += stride) d[i] = __float2bfloat16(s[i]);
}

__global__ void k_zero_u32(uint32_t* __restrict__ p, int n) {
  int i = blockIdx.x * blockDim.x + threadIdx.x;
  if (i < n) p[i] = 0u;
}

// ---------------------------------------------------------------------------
// Input layer: z[b*T+t, h] = leaky(x @ Win^T + bin)*10 (f32 VALU, K=64 tiny)
// ---------------------------------------------------------------------------
__global__ __launch_bounds__(256) void k_input_layer(
    const float* __restrict__ x, const float* __restrict__ Win,
    const float* __restrict__ bin, bh* __restrict__ z) {
  int idx = blockIdx.x * 256 + threadIdx.x;   // < B*T*H
  int r = idx >> 9;                           // (b*T + t)
  int h = idx & (Hh - 1);
  const float4* xp = reinterpret_cast<const float4*>(x + (size_t)r * INd);
  const float4* wp = reinterpret_cast<const float4*>(Win + (size_t)h * INd);
  float acc = 0.0f;
#pragma unroll
  for (int k = 0; k < INd / 4; ++k) {
    float4 a = xp[k], w = wp[k];
    acc += a.x * w.x + a.y * w.y + a.z * w.z + a.w * w.w;
  }
  acc += bin[h];
  acc = (acc >= 0.0f ? acc : 0.2f * acc) * 10.0f;
  z[idx] = __float2bfloat16(acc);
}

// ---------------------------------------------------------------------------
// xg0 precompute: xg0[t,n,b,:] = z[b,t,:] @ W_ih0[n]^T + (b_ih0+b_hh0)[n]
// Wave tile: 16 rows (b) x 64 cols, K=512 loop. A-row stride = T*H (b-major z).
// ---------------------------------------------------------------------------
__global__ __launch_bounds__(256) void k_xg0_gemm(
    const bh* __restrict__ z, const bh* __restrict__ Wih0,
    const float* __restrict__ b_ih0, const float* __restrict__ b_hh0,
    bh* __restrict__ xg0) {
  const int lane = threadIdx.x & 31, wave = threadIdx.x >> 5;
  const int half = lane >> 4, lc = lane & 15;
  int gw = blockIdx.x * 8 + wave;
  int cg = gw & 31;           // column group of 64
  int bt = (gw >> 5) & 3;     // b tile
  int t  = (gw >> 7) & 255;
  int n  = gw >> 15;
  int b0 = bt * 16;

  const bh* Arow = z + ((size_t)(b0 + lc) * Tt + t) * Hh;
  const bh* Wn   = Wih0 + (size_t)n * G4H * Hh;
  const bh* wrow[4];
#pragma unroll
  for (int s = 0; s < 4; ++s) wrow[s] = Wn + (size_t)(cg * 64 + s * 16 + lc) * Hh;

  f32x8 acc[4];
#pragma unroll
  for (int s = 0; s < 4; ++s) acc[s] = zero8();

  for (int k0 = 0; k0 < Hh; k0 += 32) {
    bf16x16 a = load_frag(Arow, k0, half);
#pragma unroll
    for (int s = 0; s < 4; ++s)
      acc[s] = wmma_bf16(a, load_frag(wrow[s], k0, half), acc[s]);
  }

  size_t rowBase = ((size_t)t * Nn + n) * Bb;
#pragma unroll
  for (int s = 0; s < 4; ++s) {
    int col = cg * 64 + s * 16 + lc;
    float bias = b_ih0[n * G4H + col] + b_hh0[n * G4H + col];
#pragma unroll
    for (int v = 0; v < 8; ++v) {
      int brow = b0 + v + 8 * half;
      xg0[(rowBase + brow) * (size_t)G4H + col] = __float2bfloat16(acc[s][v] + bias);
    }
  }
}

// ---------------------------------------------------------------------------
// LSTM layer-0 cell at step t: g = xg0[t] + h0_prev @ W_hh0^T, then cell.
// Each wave owns 16 b-rows x 16 hidden cols, computing the 4 gate tiles
// (i/f/g/o at stride 512) so the nonlinearity runs in the epilogue.
// ---------------------------------------------------------------------------
__global__ __launch_bounds__(256) void k_lstm_cell0(
    int t, const bh* __restrict__ h_prev, bh* __restrict__ h_cur,
    float* __restrict__ c, const bh* __restrict__ xg0,
    const bh* __restrict__ Whh0) {
  const int lane = threadIdx.x & 31, wave = threadIdx.x >> 5;
  const int half = lane >> 4, lc = lane & 15;
  int gw = blockIdx.x * 8 + wave;
  int ct = gw & 31;        // hidden col tile
  int bt = (gw >> 5) & 3;  // b tile
  int n  = gw >> 7;
  int b0 = bt * 16;
  int hc = ct * 16 + lc;

  const bh* Arow = h_prev + (size_t)n * Bb * Hh + (size_t)(b0 + lc) * Hh;
  const bh* Wn   = Whh0 + (size_t)n * G4H * Hh;
  const bh* wrow[4];
#pragma unroll
  for (int q = 0; q < 4; ++q) wrow[q] = Wn + (size_t)(q * Hh + hc) * Hh;

  f32x8 acc[4];
#pragma unroll
  for (int q = 0; q < 4; ++q) acc[q] = zero8();

  for (int k0 = 0; k0 < Hh; k0 += 32) {
    bf16x16 a = load_frag(Arow, k0, half);
#pragma unroll
    for (int q = 0; q < 4; ++q)
      acc[q] = wmma_bf16(a, load_frag(wrow[q], k0, half), acc[q]);
  }

  size_t xgRow = ((size_t)t * Nn + n) * Bb;
#pragma unroll
  for (int v = 0; v < 8; ++v) {
    int brow = b0 + v + 8 * half;
    size_t xb = (xgRow + brow) * (size_t)G4H;
    float gi = acc[0][v] + __bfloat162float(xg0[xb + 0 * Hh + hc]);
    float gf = acc[1][v] + __bfloat162float(xg0[xb + 1 * Hh + hc]);
    float gg = acc[2][v] + __bfloat162float(xg0[xb + 2 * Hh + hc]);
    float go = acc[3][v] + __bfloat162float(xg0[xb + 3 * Hh + hc]);
    size_t si = (size_t)n * Bb * Hh + (size_t)brow * Hh + hc;
    float cold = c[si];
    float cn = sigm(gf) * cold + sigm(gi) * tanhf(gg);
    float hn = sigm(go) * tanhf(cn);
    c[si] = cn;
    h_cur[si] = __float2bfloat16(hn);
  }
}

// ---------------------------------------------------------------------------
// LSTM layer-1 cell at step t: g = h0 @ W_ih1^T + h1_prev @ W_hh1^T + biases.
// Writes h1 state and y[n,b,t,:] (bf16, head-GEMM-friendly layout).
// ---------------------------------------------------------------------------
__global__ __launch_bounds__(256) void k_lstm_cell1(
    int t, const bh* __restrict__ h0, const bh* __restrict__ h1_prev,
    bh* __restrict__ h1_cur, float* __restrict__ c,
    const bh* __restrict__ Wih1, const bh* __restrict__ Whh1,
    const float* __restrict__ b_ih1, const float* __restrict__ b_hh1,
    bh* __restrict__ y) {
  const int lane = threadIdx.x & 31, wave = threadIdx.x >> 5;
  const int half = lane >> 4, lc = lane & 15;
  int gw = blockIdx.x * 8 + wave;
  int ct = gw & 31;
  int bt = (gw >> 5) & 3;
  int n  = gw >> 7;
  int b0 = bt * 16;
  int hc = ct * 16 + lc;

  const bh* Arow0 = h0      + (size_t)n * Bb * Hh + (size_t)(b0 + lc) * Hh;
  const bh* Arow1 = h1_prev + (size_t)n * Bb * Hh + (size_t)(b0 + lc) * Hh;
  const bh* WnA = Wih1 + (size_t)n * G4H * Hh;
  const bh* WnB = Whh1 + (size_t)n * G4H * Hh;
  const bh* wrowA[4];
  const bh* wrowB[4];
#pragma unroll
  for (int q = 0; q < 4; ++q) {
    wrowA[q] = WnA + (size_t)(q * Hh + hc) * Hh;
    wrowB[q] = WnB + (size_t)(q * Hh + hc) * Hh;
  }

  f32x8 acc[4];
#pragma unroll
  for (int q = 0; q < 4; ++q) acc[q] = zero8();

  for (int k0 = 0; k0 < Hh; k0 += 32) {
    bf16x16 a = load_frag(Arow0, k0, half);
#pragma unroll
    for (int q = 0; q < 4; ++q)
      acc[q] = wmma_bf16(a, load_frag(wrowA[q], k0, half), acc[q]);
  }
  for (int k0 = 0; k0 < Hh; k0 += 32) {
    bf16x16 a = load_frag(Arow1, k0, half);
#pragma unroll
    for (int q = 0; q < 4; ++q)
      acc[q] = wmma_bf16(a, load_frag(wrowB[q], k0, half), acc[q]);
  }

  float bias[4];
#pragma unroll
  for (int q = 0; q < 4; ++q)
    bias[q] = b_ih1[n * G4H + q * Hh + hc] + b_hh1[n * G4H + q * Hh + hc];

#pragma unroll
  for (int v = 0; v < 8; ++v) {
    int brow = b0 + v + 8 * half;
    float gi = acc[0][v] + bias[0];
    float gf = acc[1][v] + bias[1];
    float gg = acc[2][v] + bias[2];
    float go = acc[3][v] + bias[3];
    size_t si = (size_t)n * Bb * Hh + (size_t)brow * Hh + hc;
    float cold = c[si];
    float cn = sigm(gf) * cold + sigm(gi) * tanhf(gg);
    float hn = sigm(go) * tanhf(cn);
    c[si] = cn;
    bh hb = __float2bfloat16(hn);
    h1_cur[si] = hb;
    y[(((size_t)(n * Bb + brow)) * Tt + t) * Hh + hc] = hb;
  }
}

// ---------------------------------------------------------------------------
// Fused heads: H1 = leaky(y @ Wh1^T + bh1) staged to LDS (bf16), then
// out = H1 @ Wh2^T + bh2 (O=7, VALU). One block per 16-row strip per n.
// ---------------------------------------------------------------------------
__global__ __launch_bounds__(256) void k_heads(
    const bh* __restrict__ y, const bh* __restrict__ Wh1,
    const float* __restrict__ bh1, const float* __restrict__ Wh2,
    const float* __restrict__ bh2, float* __restrict__ out) {
  __shared__ bh sh[16][Mm]; // 16 KB
  const int lane = threadIdx.x & 31, wave = threadIdx.x >> 5;
  const int half = lane >> 4, lc = lane & 15;
  int n  = blockIdx.x >> 10;
  int rt = blockIdx.x & 1023;
  size_t g0 = (size_t)n * (Bb * Tt) + (size_t)rt * 16;

  const bh* Arow = y + (g0 + lc) * (size_t)Hh;
  const bh* Wn = Wh1 + (size_t)n * Mm * Hh;
  const bh* wrow[4];
#pragma unroll
  for (int s = 0; s < 4; ++s) wrow[s] = Wn + (size_t)(wave * 64 + s * 16 + lc) * Hh;

  f32x8 acc[4];
#pragma unroll
  for (int s = 0; s < 4; ++s) acc[s] = zero8();

  for (int k0 = 0; k0 < Hh; k0 += 32) {
    bf16x16 a = load_frag(Arow, k0, half);
#pragma unroll
    for (int s = 0; s < 4; ++s)
      acc[s] = wmma_bf16(a, load_frag(wrow[s], k0, half), acc[s]);
  }

#pragma unroll
  for (int s = 0; s < 4; ++s) {
    int col = wave * 64 + s * 16 + lc;
    float bias = bh1[n * Mm + col];
#pragma unroll
    for (int v = 0; v < 8; ++v) {
      int m = v + 8 * half;
      float val = acc[s][v] + bias;
      val = (val >= 0.0f ? val : 0.2f * val);
      sh[m][col] = __float2bfloat16(val);
    }
  }
  __syncthreads();

  int tid = threadIdx.x;
  if (tid < 16 * Oo) {
    int r = tid / Oo, o = tid % Oo;
    const float* w2 = Wh2 + ((size_t)n * Oo + o) * Mm;
    float a = bh2[n * Oo + o];
#pragma unroll 8
    for (int m = 0; m < Mm; ++m) a += __bfloat162float(sh[r][m]) * w2[m];
    out[(g0 + r) * Oo + o] = a;
  }
}

// ---------------------------------------------------------------------------
// Host orchestration. Workspace layout (~763 MB):
//   z(16M) | 4 LSTM weight sets bf16 (4x16.8M) | Wh1 bf16 (4.2M)
//   | states c0,c1,h0A,h0B,h1A,h1B (4.2M, zeroed) | xg0 bf16 (537M) | y bf16 (134M)
// ---------------------------------------------------------------------------
extern "C" void kernel_launch(void* const* d_in, const int* in_sizes, int n_in,
                              void* d_out, int out_size, void* d_ws, size_t ws_size,
                              hipStream_t stream) {
  (void)in_sizes; (void)n_in; (void)out_size; (void)ws_size;
  const float* x     = (const float*)d_in[0];
  const float* Win   = (const float*)d_in[1];
  const float* bin_  = (const float*)d_in[2];
  const float* W_ih0 = (const float*)d_in[3];
  const float* W_hh0 = (const float*)d_in[4];
  const float* b_ih0 = (const float*)d_in[5];
  const float* b_hh0 = (const float*)d_in[6];
  const float* W_ih1 = (const float*)d_in[7];
  const float* W_hh1 = (const float*)d_in[8];
  const float* b_ih1 = (const float*)d_in[9];
  const float* b_hh1 = (const float*)d_in[10];
  const float* Wh1   = (const float*)d_in[11];
  const float* bh1   = (const float*)d_in[12];
  const float* Wh2   = (const float*)d_in[13];
  const float* bh2   = (const float*)d_in[14];
  float* out = (float*)d_out;

  char* ws = (char*)d_ws;
  size_t off = 0;
  auto alloc = [&](size_t bytes) -> char* {
    char* p = ws + off;
    off += (bytes + 255) & ~(size_t)255;
    return p;
  };

  bh* zbf    = (bh*)alloc((size_t)Bb * Tt * Hh * 2);
  bh* Wih0bf = (bh*)alloc((size_t)Nn * G4H * Hh * 2);
  bh* Whh0bf = (bh*)alloc((size_t)Nn * G4H * Hh * 2);
  bh* Wih1bf = (bh*)alloc((size_t)Nn * G4H * Hh * 2);
  bh* Whh1bf = (bh*)alloc((size_t)Nn * G4H * Hh * 2);
  bh* Wh1bf  = (bh*)alloc((size_t)Nn * Mm * Hh * 2);
  // state block (zeroed together; all sizes are multiples of 256B -> contiguous)
  float* c0  = (float*)alloc((size_t)Nn * Bb * Hh * 4);
  float* c1  = (float*)alloc((size_t)Nn * Bb * Hh * 4);
  bh* h0A    = (bh*)alloc((size_t)Nn * Bb * Hh * 2);
  bh* h0B    = (bh*)alloc((size_t)Nn * Bb * Hh * 2);
  bh* h1A    = (bh*)alloc((size_t)Nn * Bb * Hh * 2);
  bh* h1B    = (bh*)alloc((size_t)Nn * Bb * Hh * 2);
  bh* xg0bf  = (bh*)alloc((size_t)Tt * Nn * Bb * G4H * 2);
  bh* ybf    = (bh*)alloc((size_t)Nn * Bb * Tt * Hh * 2);

  const int wcnt = Nn * G4H * Hh;       // 8,388,608
  const int w1cnt = Nn * Mm * Hh;       // 2,097,152
  k_f32_to_bf16<<<4096, 256, 0, stream>>>(W_ih0, Wih0bf, wcnt);
  k_f32_to_bf16<<<4096, 256, 0, stream>>>(W_hh0, Whh0bf, wcnt);
  k_f32_to_bf16<<<4096, 256, 0, stream>>>(W_ih1, Wih1bf, wcnt);
  k_f32_to_bf16<<<4096, 256, 0, stream>>>(W_hh1, Whh1bf, wcnt);
  k_f32_to_bf16<<<4096, 256, 0, stream>>>(Wh1, Wh1bf, w1cnt);

  // zero c0,c1,h0A,h0B,h1A,h1B: 2*1MB + 4*0.5MB = 4MB = 1,048,576 dwords
  k_zero_u32<<<4096, 256, 0, stream>>>((uint32_t*)c0, 1048576);

  k_input_layer<<<(Bb * Tt * Hh) / 256, 256, 0, stream>>>(x, Win, bin_, zbf);
  k_xg0_gemm<<<32768, 256, 0, stream>>>(zbf, Wih0bf, b_ih0, b_hh0, xg0bf);

  bh *h0p = h0A, *h0c = h0B, *h1p = h1A, *h1c = h1B;
  for (int t = 0; t < Tt; ++t) {
    k_lstm_cell0<<<128, 256, 0, stream>>>(t, h0p, h0c, c0, xg0bf, Whh0bf);
    k_lstm_cell1<<<128, 256, 0, stream>>>(t, h0c, h1p, h1c, c1, Wih1bf, Whh1bf,
                                          b_ih1, b_hh1, ybf);
    bh* tmp = h0p; h0p = h0c; h0c = tmp;
    tmp = h1p; h1p = h1c; h1c = tmp;
  }

  k_heads<<<Nn * 1024, 256, 0, stream>>>(ybf, Wh1bf, bh1, Wh2, bh2, out);
}
